// MOSI_attention_classifier_62706522522040
// MI455X (gfx1250) — matrix operand order repaired and checked by reading, not compiled, not run on hardware
//
#include <hip/hip_runtime.h>

// ---------------------------------------------------------------------------
// MI455X (gfx1250) persistent dual-LSTM + attention, bf16 WMMA 16x16x32.
// All activations are converted to bf16 WMMA A-fragment layout ONCE at their
// producer; every GEMM inner iteration is pure b128 loads + v_wmma.
// Weights packed once to bf16 B-fragment order (~45 MB -> L2 resident).
// 512 sequential steps executed by one cooperative grid with global barriers.
// ---------------------------------------------------------------------------

typedef __attribute__((ext_vector_type(16))) __bf16 v16bf;
typedef __attribute__((ext_vector_type(8)))  float  v8f;

#define T_STEPS 512
#define NWG 32

__device__ __forceinline__ __bf16 f2bf(float f) {
  union { float f; unsigned u; } v; v.f = f;
  unsigned r = v.u + 0x7FFFu + ((v.u >> 16) & 1u);   // round to nearest even
  unsigned short h = (unsigned short)(r >> 16);
  return __builtin_bit_cast(__bf16, h);
}

__device__ __forceinline__ float sigf(float x) { return 1.f / (1.f + __expf(-x)); }

// Fragment layout (ISA 7.12.2, 16-bit A 16x32 == B 32x16 storage granule):
// frag kt is 512 bf16: [lane 0..31][elem 0..15], 32B contiguous per lane.
__device__ __forceinline__ v16bf load_frag(const __bf16* f, int kt, int lane) {
  return *(const v16bf*)(f + (size_t)kt * 512 + lane * 16);
}

// Scatter one (row m, k) activation element into A-fragment layout.
// lane = m + 16*bit3(k%32); elem = 8*bit4(k%32) + (k%8).
__device__ __forceinline__ void store_frag_elem(__bf16* frag, int kt, int m, int ko, float v) {
  const int lane = m + ((ko >> 3) & 1) * 16;
  const int e    = ((ko >> 4) << 3) + (ko & 7);
  frag[(size_t)kt * 512 + lane * 16 + e] = f2bf(v);
}

__device__ __forceinline__ v8f wmma_bf16(v16bf a, v16bf b, v8f c) {
  return __builtin_amdgcn_wmma_f32_16x16x32_bf16(false, a, false, b, (short)0, c, false, false);
}

// Sense-reversing grid barrier (all NWG workgroups resident).
__device__ __forceinline__ void grid_sync(unsigned* cnt, volatile unsigned* gen) {
  __syncthreads();
  if (threadIdx.x == 0) {
    __threadfence();
    unsigned g = *gen;
    if (atomicAdd(cnt, 1u) == (unsigned)(NWG - 1)) {
      *cnt = 0u;
      __threadfence();
      *gen = g + 1u;
    } else {
      while (*gen == g) { __builtin_amdgcn_s_sleep(1); }
      __threadfence();
    }
  }
  __syncthreads();
}

// ---------------------------------------------------------------------------
// One-time packing kernels
// ---------------------------------------------------------------------------

// f32 [Korig x N] row-major -> bf16 tiled fragments [Kpad/32][N/16][32 lanes][16].
__global__ void pack_w(const float* __restrict__ W, __bf16* __restrict__ out,
                       int Korig, int Kpad, int N) {
  const int ntiles = N >> 4;
  const int nkt    = Kpad >> 5;
  const size_t total = (size_t)nkt * ntiles * 32;
  for (size_t idx = (size_t)blockIdx.x * blockDim.x + threadIdx.x; idx < total;
       idx += (size_t)gridDim.x * blockDim.x) {
    const int lane  = (int)(idx & 31);
    const size_t fi = idx >> 5;
    const int nt = (int)(fi % ntiles);
    const int kt = (int)(fi / ntiles);
    const int n  = nt * 16 + (lane & 15);
    const int kb = kt * 32 + (lane >> 4) * 8;
    __bf16* o = out + idx * 16;
#pragma unroll
    for (int e = 0; e < 16; ++e) {
      const int k = kb + ((e < 8) ? e : (e + 8));
      o[e] = f2bf((k < Korig) ? W[(size_t)k * N + n] : 0.f);
    }
  }
}

// Inputs x [T][8][Kin] -> per-step A-fragments [T][nkt][512] (rows 8..15 = 0).
__global__ void pack_x_frags(const float* __restrict__ x, __bf16* __restrict__ out,
                             int Kin, int nkt) {
  const size_t total = (size_t)T_STEPS * nkt * 512;
  for (size_t idx = (size_t)blockIdx.x * blockDim.x + threadIdx.x; idx < total;
       idx += (size_t)gridDim.x * blockDim.x) {
    const int e    = (int)(idx & 15);
    const int lane = (int)((idx >> 4) & 31);
    const size_t fi = idx >> 9;
    const int kt = (int)(fi % nkt);
    const int t  = (int)(fi / nkt);
    const int m  = lane & 15;
    const int kh = (lane >> 4) * 8;
    const int ko = (e < 8) ? (kh + e) : (16 + kh + e - 8);
    const int k  = kt * 32 + ko;
    const float v = (m < 8 && k < Kin) ? x[((size_t)t * 8 + m) * Kin + k] : 0.f;
    out[idx] = f2bf(v);
  }
}

__global__ void init_zero(float* __restrict__ p, size_t n) {
  for (size_t i = (size_t)blockIdx.x * blockDim.x + threadIdx.x; i < n;
       i += (size_t)gridDim.x * blockDim.x)
    p[i] = 0.f;
}

// ---------------------------------------------------------------------------
// Persistent recurrent kernel
// ---------------------------------------------------------------------------

struct RnnParams {
  const __bf16 *lanWx, *lanWh, *lanWz, *faceWx, *faceWh, *faceWz, *Wa, *Wac;
  const __bf16 *xlanfrag, *xfacefrag;            // [T][nkt][512]
  const float *lan_b, *face_b, *b_a, *b_ac, *W_cout, *b_cout;
  float *hbuf, *cl, *cf, *zbuf, *logits;         // fp32 state
  __bf16 *hfrag0, *hfrag1, *zfrag0, *zfrag1, *ufrag;  // bf16 A-fragments
  unsigned *barCnt;
  volatile unsigned *barGen;
  float *out;
};

__device__ __forceinline__ v16bf load_b(const __bf16* wt, int kt, int ntiles, int nt, int lane) {
  return *(const v16bf*)(wt + (((size_t)kt * ntiles + nt) * 512) + lane * 16);
}

__global__ void __launch_bounds__(256) mosi_rnn_persistent(RnnParams p) {
  __shared__ float red[2][4][8][16];             // phase-B cross-wave reduction
  const int lane   = (int)(threadIdx.x & 31);
  const int wid    = (int)((blockIdx.x * blockDim.x + threadIdx.x) >> 5);
  const int nwaves = (int)((gridDim.x * blockDim.x) >> 5);

  for (int t = 0; t < T_STEPS; ++t) {
    const __bf16* hfragin  = (t & 1) ? p.hfrag1 : p.hfrag0;
    __bf16*       hfragout = (t & 1) ? p.hfrag0 : p.hfrag1;
    const __bf16* zfragin  = (t & 1) ? p.zfrag1 : p.zfrag0;
    __bf16*       zfragout = (t & 1) ? p.zfrag0 : p.zfrag1;

    // ---- Phase A: gate unit-tiles (0..127) + attention logit tiles (128..255)
    for (int tile = wid; tile < 256; tile += nwaves) {
      if (tile < 128) {
        const int isFace = (tile >= 64);
        const int ut = tile & 63;                      // 16 hidden units
        const __bf16* xfr = isFace ? (p.xfacefrag + (size_t)t * 3 * 512)
                                   : (p.xlanfrag  + (size_t)t * 10 * 512);
        const int     nktx = isFace ? 3 : 10;
        const __bf16* Wx = isFace ? p.faceWx : p.lanWx;
        const __bf16* Wh = isFace ? p.faceWh : p.lanWh;
        const __bf16* Wz = isFace ? p.faceWz : p.lanWz;
        const __bf16* hfr = hfragin + (size_t)(isFace ? 32 : 0) * 512;
        __bf16*       hfo = hfragout + (size_t)(isFace ? 32 : 0) * 512;
        const float*  bias = isFace ? p.face_b : p.lan_b;
        float*        cbuf = isFace ? p.cf : p.cl;
        const int  cellOff = isFace ? 1024 : 0;

        v8f acc[4] = {};                                // i, f, g, o
        for (int kt = 0; kt < nktx; ++kt) {             // x @ Wx
          v16bf a = load_frag(xfr, kt, lane);
#pragma unroll
          for (int g = 0; g < 4; ++g)
            acc[g] = wmma_bf16(a, load_b(Wx, kt, 256, g * 64 + ut, lane), acc[g]);
        }
        for (int kt = 0; kt < 32; ++kt) {               // h @ Wh
          v16bf a = load_frag(hfr, kt, lane);
#pragma unroll
          for (int g = 0; g < 4; ++g)
            acc[g] = wmma_bf16(a, load_b(Wh, kt, 256, g * 64 + ut, lane), acc[g]);
        }
        for (int kt = 0; kt < 32; ++kt) {               // z @ Wz
          v16bf a = load_frag(zfragin, kt, lane);
#pragma unroll
          for (int g = 0; g < 4; ++g)
            acc[g] = wmma_bf16(a, load_b(Wz, kt, 256, g * 64 + ut, lane), acc[g]);
        }
        // LSTM nonlinearity: lanes 0-15 hold batch rows 0..7 (VGPR r = row r).
        if (lane < 16) {
          const int unit = ut * 16 + lane;
          const float bi = bias[unit],        bfv = bias[1024 + unit];
          const float bg = bias[2048 + unit], bo  = bias[3072 + unit];
          const int kt = unit >> 5, ko = unit & 31;
#pragma unroll
          for (int r = 0; r < 8; ++r) {
            const float iv = sigf(acc[0][r] + bi);
            const float fv = sigf(acc[1][r] + bfv);
            const float gv = tanhf(acc[2][r] + bg);
            const float ov = sigf(acc[3][r] + bo);
            const float cn = fv * cbuf[r * 1024 + unit] + iv * gv;
            cbuf[r * 1024 + unit] = cn;
            const float hv = ov * tanhf(cn);
            p.hbuf[r * 2048 + cellOff + unit] = hv;       // fp32 for softmax stage
            store_frag_elem(hfo, kt, r, ko, hv);          // bf16 for next-step GEMM
          }
        }
      } else {                                           // attention logits
        const int nt = tile - 128;
        v8f acc = {};
        for (int kt = 0; kt < 32; ++kt) {
          v16bf a = load_frag(zfragin, kt, lane);
          acc = wmma_bf16(a, load_b(p.Wa, kt, 128, nt, lane), acc);
        }
        if (lane < 16) {
          const int col = nt * 16 + lane;
          const float bb = p.b_a[col];
#pragma unroll
          for (int r = 0; r < 8; ++r) p.logits[r * 2048 + col] = acc[r] + bb;
        }
      }
    }
    grid_sync(p.barCnt, p.barGen);

    // ---- Phase A2: softmax + u = h*attn, emitted directly as bf16 fragments
    if (blockIdx.x == 0) {
      const int wv = (int)(threadIdx.x >> 5);
      if (wv < 8) {
        const float* L = p.logits + wv * 2048;
        float mx = -3.4e38f;
        for (int j = lane; j < 2048; j += 32) mx = fmaxf(mx, L[j]);
#pragma unroll
        for (int o = 16; o; o >>= 1) mx = fmaxf(mx, __shfl_xor(mx, o));
        float s = 0.f;
        for (int j = lane; j < 2048; j += 32) s += __expf(L[j] - mx);
#pragma unroll
        for (int o = 16; o; o >>= 1) s += __shfl_xor(s, o);
        const float inv = 1.f / s;
        const float* hr = p.hbuf + wv * 2048;
        for (int j = lane; j < 2048; j += 32) {
          const float uv = hr[j] * __expf(L[j] - mx) * inv;
          store_frag_elem(p.ufrag, j >> 5, wv, j & 31, uv);
        }
      }
    }
    grid_sync(p.barCnt, p.barGen);

    // ---- Phase B: z_new = u @ W_ac + b_ac. 4 waves per tile split K, LDS
    //      reduce, slice-0 wave finalizes fp32 z + bf16 z-fragments.
    {
      const int bw    = (int)(threadIdx.x >> 5);
      const int tileL = bw >> 2;                          // 0..1
      const int nt    = (int)blockIdx.x * 2 + tileL;      // 0..63
      const int ks    = bw & 3;                           // K slice
      v8f acc = {};
      for (int kt = ks * 16; kt < ks * 16 + 16; ++kt) {
        v16bf a = load_frag(p.ufrag, kt, lane);
        acc = wmma_bf16(a, load_b(p.Wac, kt, 64, nt, lane), acc);
      }
      if (lane < 16) {
#pragma unroll
        for (int r = 0; r < 8; ++r) red[tileL][ks][r][lane] = acc[r];
      }
      __syncthreads();
      if (ks == 0 && lane < 16) {
        const int col = nt * 16 + lane;
        const float bb = p.b_ac[col];
        const int kt = col >> 5, ko = col & 31;
#pragma unroll
        for (int r = 0; r < 8; ++r) {
          const float zv = red[tileL][0][r][lane] + red[tileL][1][r][lane] +
                           red[tileL][2][r][lane] + red[tileL][3][r][lane] + bb;
          p.zbuf[r * 1024 + col] = zv;                    // fp32 for classifier
          store_frag_elem(zfragout, kt, r, ko, zv);       // bf16 for next step
        }
      }
    }
    grid_sync(p.barCnt, p.barGen);
  }

  // ---- Final classifier: out[b] = z_final[b] . W_cout + b_cout
  if (blockIdx.x == 0) {
    const int wv = (int)(threadIdx.x >> 5);
    if (wv < 8) {
      float s = 0.f;
      const float* zr = p.zbuf + wv * 1024;
      for (int j = lane; j < 1024; j += 32) s += zr[j] * p.W_cout[j];
#pragma unroll
      for (int o = 16; o; o >>= 1) s += __shfl_xor(s, o);
      if (lane == 0) p.out[wv] = s + p.b_cout[0];
    }
  }
}

// ---------------------------------------------------------------------------
// Host launcher
// ---------------------------------------------------------------------------

extern "C" void kernel_launch(void* const* d_in, const int* in_sizes, int n_in,
                              void* d_out, int out_size, void* d_ws, size_t ws_size,
                              hipStream_t stream) {
  (void)in_sizes; (void)n_in; (void)out_size; (void)ws_size;
  const float* x_lan   = (const float*)d_in[0];
  const float* x_face  = (const float*)d_in[1];
  const float* lan_Wx  = (const float*)d_in[2];
  const float* lan_Wh  = (const float*)d_in[3];
  const float* lan_Wz  = (const float*)d_in[4];
  const float* lan_b   = (const float*)d_in[5];
  const float* face_Wx = (const float*)d_in[6];
  const float* face_Wh = (const float*)d_in[7];
  const float* face_Wz = (const float*)d_in[8];
  const float* face_b  = (const float*)d_in[9];
  const float* W_a     = (const float*)d_in[10];
  const float* b_a     = (const float*)d_in[11];
  const float* W_ac    = (const float*)d_in[12];
  const float* b_ac    = (const float*)d_in[13];
  const float* W_cout  = (const float*)d_in[14];
  const float* b_cout  = (const float*)d_in[15];

  char* w = (char*)d_ws;
  size_t off = 0;
  auto take = [&](size_t bytes) -> void* {
    void* p = w + off;
    off = (off + bytes + 255) & ~(size_t)255;
    return p;
  };

  // bf16 tiled weights (B-fragment order)
  __bf16* lanWx_t  = (__bf16*)take((size_t)10 * 256 * 512 * 2);  // K 300->320
  __bf16* lanWh_t  = (__bf16*)take((size_t)32 * 256 * 512 * 2);
  __bf16* lanWz_t  = (__bf16*)take((size_t)32 * 256 * 512 * 2);
  __bf16* faceWx_t = (__bf16*)take((size_t)3  * 256 * 512 * 2);  // K 74->96
  __bf16* faceWh_t = (__bf16*)take((size_t)32 * 256 * 512 * 2);
  __bf16* faceWz_t = (__bf16*)take((size_t)32 * 256 * 512 * 2);
  __bf16* Wa_t     = (__bf16*)take((size_t)32 * 128 * 512 * 2);
  __bf16* Wac_t    = (__bf16*)take((size_t)64 * 64  * 512 * 2);
  // input A-fragments for every step
  __bf16* xlanfrag  = (__bf16*)take((size_t)T_STEPS * 10 * 512 * 2);
  __bf16* xfacefrag = (__bf16*)take((size_t)T_STEPS * 3 * 512 * 2);
  // state (contiguous, zeroed each launch; fragment rows 8..15 stay zero)
  float* stateBase = (float*)(w + off);
  float* hbuf   = (float*)take(16384 * 4);            // [8][2048] h fp32
  float* cl     = (float*)take(8192 * 4);             // [8][1024]
  float* cf     = (float*)take(8192 * 4);
  float* zbuf   = (float*)take(8192 * 4);             // [8][1024] z fp32
  float* logits = (float*)take(16384 * 4);            // [8][2048]
  __bf16* hfrag0 = (__bf16*)take((size_t)64 * 512 * 2);  // [2 cells][32][512]
  __bf16* hfrag1 = (__bf16*)take((size_t)64 * 512 * 2);
  __bf16* zfrag0 = (__bf16*)take((size_t)32 * 512 * 2);  // [32][512]
  __bf16* zfrag1 = (__bf16*)take((size_t)32 * 512 * 2);
  __bf16* ufrag  = (__bf16*)take((size_t)64 * 512 * 2);  // [64][512]
  unsigned* bar  = (unsigned*)take(64);
  const size_t stateFloats = (16384 + 8192 + 8192 + 8192 + 16384) +
                             (16384 + 16384 + 8192 + 8192 + 16384) + 16;

  // One-time (per launch) packing + init, serialized on `stream`.
  pack_w<<<512, 256, 0, stream>>>(lan_Wx,  lanWx_t,  300,  320,  4096);
  pack_w<<<512, 256, 0, stream>>>(lan_Wh,  lanWh_t,  1024, 1024, 4096);
  pack_w<<<512, 256, 0, stream>>>(lan_Wz,  lanWz_t,  1024, 1024, 4096);
  pack_w<<<512, 256, 0, stream>>>(face_Wx, faceWx_t, 74,   96,   4096);
  pack_w<<<512, 256, 0, stream>>>(face_Wh, faceWh_t, 1024, 1024, 4096);
  pack_w<<<512, 256, 0, stream>>>(face_Wz, faceWz_t, 1024, 1024, 4096);
  pack_w<<<512, 256, 0, stream>>>(W_a,     Wa_t,     1024, 1024, 2048);
  pack_w<<<512, 256, 0, stream>>>(W_ac,    Wac_t,    2048, 2048, 1024);
  pack_x_frags<<<512, 256, 0, stream>>>(x_lan,  xlanfrag,  300, 10);
  pack_x_frags<<<512, 256, 0, stream>>>(x_face, xfacefrag, 74,  3);
  init_zero<<<128, 256, 0, stream>>>(stateBase, stateFloats);

  RnnParams p;
  p.lanWx = lanWx_t;  p.lanWh = lanWh_t;  p.lanWz = lanWz_t;
  p.faceWx = faceWx_t; p.faceWh = faceWh_t; p.faceWz = faceWz_t;
  p.Wa = Wa_t; p.Wac = Wac_t;
  p.xlanfrag = xlanfrag; p.xfacefrag = xfacefrag;
  p.lan_b = lan_b; p.face_b = face_b; p.b_a = b_a; p.b_ac = b_ac;
  p.W_cout = W_cout; p.b_cout = b_cout;
  p.hbuf = hbuf; p.cl = cl; p.cf = cf; p.zbuf = zbuf; p.logits = logits;
  p.hfrag0 = hfrag0; p.hfrag1 = hfrag1; p.zfrag0 = zfrag0; p.zfrag1 = zfrag1;
  p.ufrag = ufrag;
  p.barCnt = bar; p.barGen = (volatile unsigned*)(bar + 1);
  p.out = (float*)d_out;

  mosi_rnn_persistent<<<NWG, 256, 0, stream>>>(p);
}
